// VolumeFeatureRenderer_12034498363866
// MI455X (gfx1250) — matrix operand order repaired and checked by reading, not compiled, not run on hardware
//
#include <hip/hip_runtime.h>
#include <hip/hip_bf16.h>
#include <math.h>

typedef __attribute__((ext_vector_type(16))) _Float16 v16h;
typedef __attribute__((ext_vector_type(8)))  _Float16 v8h;
typedef __attribute__((ext_vector_type(8)))  float    v8f;

#define SR      296      // activation row stride in f16 elements (288 + 8 pad)
#define NPTS    96       // points per workgroup (4 rays x 24 samples)
#define NTHREADS 192     // 6 wave32, each wave owns 16 rows

// ---------------- ws (scratch) layout ----------------
// f16 region:
//   [0 .. 8192)            w0 padded to [256][32]
//   [8192 .. 466944)       ws  7 x [256][256]
//   [466944 .. 540672)     view_w padded to [256][288]
// f32 region (byte offset 1081344):
//   gammaArr [2][9][256], betaArr [2][9][256]
#define W0H_ELEMS   8192
#define WSH_OFF     8192
#define VWH_OFF     466944
#define F16_TOTAL   540672
#define GB_BYTE_OFF 1081344

// LDS partition sizes (bytes)
#define WBUF_BYTES  147456   // max layer weight: view 256x288 f16
#define ACT_BYTES   (NPTS * SR * 2)

// ================= prep kernel 1: f16 weight conversion (with padding) =================
__global__ void prep_weights(const float* __restrict__ w0,
                             const float* __restrict__ ws7,
                             const float* __restrict__ vw,
                             _Float16* __restrict__ out)
{
    int idx = blockIdx.x * blockDim.x + threadIdx.x;
    if (idx >= F16_TOTAL) return;
    float v;
    if (idx < W0H_ELEMS) {                       // w0: [256][3] -> [256][32]
        int o = idx >> 5, k = idx & 31;
        v = (k < 3) ? w0[o * 3 + k] : 0.0f;
    } else if (idx < VWH_OFF) {                  // ws: direct copy
        v = ws7[idx - WSH_OFF];
    } else {                                     // view_w: [256][259] -> [256][288]
        int r = idx - VWH_OFF;
        int o = r / 288, k = r - o * 288;
        v = (k < 259) ? vw[o * 259 + k] : 0.0f;
    }
    out[idx] = (_Float16)v;
}

// ================= prep kernel 2: FiLM gamma/beta per (batch, layer, unit) =================
__global__ void prep_gb(const float* __restrict__ styles,
                        const float* __restrict__ gamma_w, const float* __restrict__ gamma_b,
                        const float* __restrict__ beta_w,  const float* __restrict__ beta_b,
                        const float* __restrict__ view_gw, const float* __restrict__ view_gb,
                        const float* __restrict__ view_bw, const float* __restrict__ view_bb,
                        float* __restrict__ gOut, float* __restrict__ bOut)
{
    int idx = blockIdx.x * blockDim.x + threadIdx.x;
    if (idx >= 2 * 9 * 256) return;
    int o = idx & 255;
    int l = (idx >> 8) % 9;
    int b = idx / (9 * 256);
    const float* st = styles + b * 256;
    const float *gw, *bw;
    float gb, bb;
    if (l < 8) {
        gw = gamma_w + (l * 256 + o) * 256; gb = gamma_b[l * 256 + o];
        bw = beta_w  + (l * 256 + o) * 256; bb = beta_b [l * 256 + o];
    } else {
        gw = view_gw + o * 256; gb = view_gb[o];
        bw = view_bw + o * 256; bb = view_bb[o];
    }
    float ga = 0.0f, ba = 0.0f;
    for (int s = 0; s < 256; ++s) { ga += st[s] * gw[s]; ba += st[s] * bw[s]; }
    gOut[idx] = 15.0f * (ga + gb) + 30.0f;   // LinearLayer(std=15, bias 30)
    bOut[idx] = 0.25f * (ba + bb);           // LinearLayer(std=0.25)
}

// ================= Tensor Data Mover: flat global->LDS weight DMA =================
#if __has_builtin(__builtin_amdgcn_tensor_load_to_lds)
#define HAVE_TDM 1
typedef __attribute__((ext_vector_type(4))) unsigned int u32x4;
typedef __attribute__((ext_vector_type(8))) int i32x8;
typedef __attribute__((ext_vector_type(4))) int i32x4;

// 1-D copy of `bytes` (multiple of 8) from gsrc to LDS, per ISA D# spec:
//  group0: count=1 | lds_addr | global_addr[56:0] | type=2
//  group1: data_size=8B, tensor_dim0 = tile_dim0 = bytes/8 (<= 18432 fits 16-bit tile dim)
// This toolchain exposes the 6-arg builtin (extra int32x8 group before cpol).
__device__ __forceinline__ void tdm_issue(const _Float16* __restrict__ gsrc,
                                          void* ldst, unsigned bytes)
{
    unsigned n8 = bytes >> 3;                               // 8-byte units
    unsigned long long ga = (unsigned long long)(uintptr_t)gsrc;
    unsigned lds = (unsigned)(uintptr_t)ldst;               // low 32 bits = LDS offset
    u32x4 g0 = { 1u,                                        // count=1 (user descriptor)
                 lds,                                       // lds_addr [63:32]
                 (unsigned)ga,                              // global_addr[31:0]
                 (unsigned)((ga >> 32) & 0x01FFFFFFull)     // global_addr[56:32]
                   | 0x80000000u };                         // type=2 ("image")
    i32x8 g1 = { 0x00030000,                                // data_size=3 (8B), mask=0
                 (int)((n8 & 0xFFFFu) << 16),               // tensor_dim0[15:0] @ bits 63:48
                 (int)(n8 >> 16),                           // tensor_dim0[31:16]; tensor_dim1=0
                 (int)(n8 << 16),                           // tile_dim0 @ bits 127:112
                 0,                                         // tile_dim1=0 (1-D), tile_dim2=0
                 (int)n8,                                   // tensor_dim0_stride[31:0]
                 0, 0 };
    i32x4 z4 = { 0, 0, 0, 0 };
    i32x8 z8 = { 0, 0, 0, 0, 0, 0, 0, 0 };
    __builtin_amdgcn_tensor_load_to_lds(g0, g1, z4, z4, z8, 0);
}
#endif

// ================= FiLM-SIREN layer via WMMA (A and B both from LDS) =================
__device__ __forceinline__ void film_layer(
    const _Float16* __restrict__ actIn, _Float16* __restrict__ actOut,
    const _Float16* __restrict__ wgt, int KP, int nkt,
    const float* __restrict__ bias,
    const float* __restrict__ gArr, const float* __restrict__ bArr,
    int lane, int rbase)
{
    const int half = lane >> 4;      // 0: K chunks {0..7,16..23}; 1: {8..15,24..31}
    const int m16  = lane & 15;
    const _Float16* aRow = actIn + (rbase + m16) * SR + half * 8;
    const int r0 = rbase + half * 8; // D rows this lane holds (VGPR j -> row r0+j)

    for (int ct = 0; ct < 16; ++ct) {
        const int col = ct * 16 + m16;                        // N index for this lane
        const _Float16* bPtr = wgt + col * KP + half * 8;     // B[k][n] = W[n][k]
        v8f acc = {};
        for (int kt = 0; kt < nkt; ++kt) {
            v8h a0 = *(const v8h*)(aRow + kt * 32);           // K = 32kt + 8h + 0..7
            v8h a1 = *(const v8h*)(aRow + kt * 32 + 16);      // K = 32kt + 16 + 8h + 0..7
            v16h A = __builtin_shufflevector(a0, a1,
                     0,1,2,3,4,5,6,7,8,9,10,11,12,13,14,15);
            v8h b0 = *(const v8h*)(bPtr + kt * 32);           // K = 32kt + 8h + 0..7
            v8h b1 = *(const v8h*)(bPtr + kt * 32 + 16);      // K = 32kt + 16 + 8h + 0..7
            v16h B = __builtin_shufflevector(b0, b1,
                     0,1,2,3,4,5,6,7,8,9,10,11,12,13,14,15);
            acc = __builtin_amdgcn_wmma_f32_16x16x32_f16(
                      false, A, false, B, (short)0, acc, false, false);
        }
        const float lb = bias[col];
        const float gm = gArr[col];
        const float bt = bArr[col];
#pragma unroll
        for (int j = 0; j < 8; ++j) {
            float v = __sinf(gm * (acc[j] + lb) + bt);        // sin(gamma*(xW^T+b)+beta)
            actOut[(r0 + j) * SR + col] = (_Float16)v;
        }
    }
}

// ================= fused renderer =================
__global__ __launch_bounds__(NTHREADS)
void render_kernel(
    const float* __restrict__ cam_poses, const float* __restrict__ focal,
    const float* __restrict__ nearv,     const float* __restrict__ farv,
    const float* __restrict__ b0,        const float* __restrict__ bs,
    const float* __restrict__ view_b,
    const float* __restrict__ rgb_w,     const float* __restrict__ rgb_b,
    const float* __restrict__ sigma_w,   const float* __restrict__ sigma_b,
    const float* __restrict__ sigmoid_beta,
    const _Float16* __restrict__ w0h, const _Float16* __restrict__ wsh,
    const _Float16* __restrict__ vwh,
    const float* __restrict__ gammaArr, const float* __restrict__ betaArr,
    float* __restrict__ out)
{
    extern __shared__ char smem[];
    _Float16* wbuf = (_Float16*)smem;                    // [<=147456 B] staged weights
    _Float16* buf0 = (_Float16*)(smem + WBUF_BYTES);     // [96][SR]
    _Float16* buf1 = buf0 + NPTS * SR;                   // [96][SR]
    float* distsL = (float*)((char*)buf1 + ACT_BYTES);   // [96]
    float* sdfL   = distsL + NPTS;                       // [96]
    float* rgbL   = sdfL + NPTS;                         // [96][3]  sigmoid(rgb)
    float* wL     = rgbL + NPTS * 3;                     // [96]     volume weights

    const int tid = threadIdx.x;
    const int wg  = blockIdx.x;
    const int bwg = wg >> 10;                            // 1024 workgroups per batch

    // kick layer-0 weight DMA immediately; it overlaps the ray setup below
#if HAVE_TDM
    if (tid < 32) tdm_issue(w0h, wbuf, 32 * 256 * 2);
#endif

    // ---- setup: rays, sample points, viewdirs, dists ----
    if (tid < NPTS) {
        int p = tid;
        int q = p / 24, s = p - q * 24;
        int ray = wg * 4 + q;
        int bb  = ray >> 12;
        int rem = ray & 4095;
        int hi = rem >> 6, wi = rem & 63;
        float f  = focal[bb], nr = nearv[bb], fr = farv[bb];
        const float* P = cam_poses + bb * 12;            // [3][4] row-major
        float d0 = ((float)wi + 0.5f - 32.0f) / f;
        float d1 = -(((float)hi + 0.5f) - 32.0f) / f;
        float d2 = -1.0f;
        float rd0 = d0*P[0] + d1*P[1] + d2*P[2];
        float rd1 = d0*P[4] + d1*P[5] + d2*P[6];
        float rd2 = d0*P[8] + d1*P[9] + d2*P[10];
        float ro0 = P[3], ro1 = P[7], ro2 = P[11];
        float nrm = sqrtf(rd0*rd0 + rd1*rd1 + rd2*rd2);
        float ts = (float)s * (1.0f / 24.0f);            // linspace(0, 1-1/NS, NS)
        float z  = nr * (1.0f - ts) + fr * ts;
        float sc = 2.0f / (fr - nr);                     // z_normalize
        _Float16* row = buf0 + p * SR;
        row[0] = (_Float16)((ro0 + rd0 * z) * sc);
        row[1] = (_Float16)((ro1 + rd1 * z) * sc);
        row[2] = (_Float16)((ro2 + rd2 * z) * sc);
        for (int k = 3; k < 32; ++k) row[k] = (_Float16)0.0f;
        float inn = 1.0f / nrm;                          // viewdirs -> view-input cols
        row[256] = (_Float16)(rd0 * inn);
        row[257] = (_Float16)(rd1 * inn);
        row[258] = (_Float16)(rd2 * inn);
        for (int k = 259; k < 288; ++k) row[k] = (_Float16)0.0f;
        distsL[p] = ((s < 23) ? (fr - nr) * (1.0f / 24.0f) : 1e10f) * nrm;
    }
#if HAVE_TDM
    if (tid < 32) __builtin_amdgcn_s_wait_tensorcnt(0);
#else
    for (int e = tid * 8; e < 32 * 256; e += NTHREADS * 8)
        *(v8h*)(wbuf + e) = *(const v8h*)(w0h + e);
#endif
    __syncthreads();

    const int lane = tid & 31, wv = tid >> 5, rbase = wv * 16;
    const float* gB = gammaArr + bwg * 9 * 256;
    const float* eB = betaArr  + bwg * 9 * 256;

    // layer 0: 3(->32) -> 256, in buf0, out buf1
    film_layer(buf0, buf1, wbuf, 32, 1, b0, gB, eB, lane, rbase);

    // hidden layers 1..7: 256 -> 256 (ping-pong; final h lands in buf0)
    for (int l = 0; l < 7; ++l) {
        const _Float16* actIn  = (l & 1) ? buf0 : buf1;
        _Float16*       actOut = (l & 1) ? buf1 : buf0;
        __syncthreads();                                 // wbuf free for next layer
        __builtin_prefetch(wsh + l * 65536 + tid * 64, 0, 1);  // warm GL2 for DMA
#if HAVE_TDM
        if (tid < 32) {
            tdm_issue(wsh + l * 65536, wbuf, 256 * 256 * 2);
            __builtin_amdgcn_s_wait_tensorcnt(0);
        }
#else
        for (int e = tid * 8; e < 256 * 256; e += NTHREADS * 8)
            *(v8h*)(wbuf + e) = *(const v8h*)(wsh + l * 65536 + e);
#endif
        __syncthreads();
        film_layer(actIn, actOut, wbuf, 256, 8,
                   bs + l * 256, gB + (l + 1) * 256, eB + (l + 1) * 256, lane, rbase);
    }
    __syncthreads();                                     // layer 7 done; wbuf free

    // view-layer weight DMA overlaps the sigma head below
#if HAVE_TDM
    if (tid < 32) tdm_issue(vwh, wbuf, 256 * 288 * 2);
#endif

    // sigma head: sdf = h . sigma_w + sigma_b   (h = buf0 rows)
    if (tid < NPTS) {
        const _Float16* row = buf0 + tid * SR;
        float acc = 0.0f;
        for (int c = 0; c < 256; ++c) acc += (float)row[c] * sigma_w[c];
        sdfL[tid] = acc + sigma_b[0];
    }
#if HAVE_TDM
    if (tid < 32) __builtin_amdgcn_s_wait_tensorcnt(0);
#else
    for (int e = tid * 8; e < 256 * 288; e += NTHREADS * 8)
        *(v8h*)(wbuf + e) = *(const v8h*)(vwh + e);
#endif
    __syncthreads();

    // view layer: [h | viewdir] (259 -> pad 288) -> 256, in buf0, out buf1 (= feat)
    film_layer(buf0, buf1, wbuf, 288, 9, view_b, gB + 8 * 256, eB + 8 * 256, lane, rbase);
    __syncthreads();

    // rgb head: sigmoid(feat . rgb_w + rgb_b)
    if (tid < NPTS) {
        const _Float16* row = buf1 + tid * SR;
        float a0 = 0.0f, a1 = 0.0f, a2 = 0.0f;
        for (int c = 0; c < 256; ++c) {
            float hv = (float)row[c];
            a0 += hv * rgb_w[c];
            a1 += hv * rgb_w[256 + c];
            a2 += hv * rgb_w[512 + c];
        }
        rgbL[tid * 3 + 0] = 1.0f / (1.0f + __expf(-(a0 + rgb_b[0])));
        rgbL[tid * 3 + 1] = 1.0f / (1.0f + __expf(-(a1 + rgb_b[1])));
        rgbL[tid * 3 + 2] = 1.0f / (1.0f + __expf(-(a2 + rgb_b[2])));
    }
    __syncthreads();

    // volume integration weights (one thread per ray), with force_background
    if (tid < 4) {
        const float sb = sigmoid_beta[0];
        float vis = 1.0f, wsum = 0.0f;
        int base = tid * 24;
        for (int s = 0; s < 24; ++s) {
            float sd = sdfL[base + s];
            float sg = (1.0f / (1.0f + __expf(sd / sb))) / sb;   // sigmoid(-sdf/b)/b
            float al = 1.0f - __expf(-sg * distsL[base + s]);
            float w  = al * vis;
            if (s < 23) { wL[base + s] = w; wsum += w; }
            vis *= (1.0f - al + 1e-10f);
        }
        wL[base + 23] = 1.0f - wsum;                             // bg absorbs remainder
    }
    __syncthreads();

    // outputs: [ray][0..2] = -1 + 2*sum(w*sigmoid(rgb)); [ray][3..258] = sum(w*feat)
    for (int idx = tid; idx < 4 * 259; idx += NTHREADS) {
        int q = idx / 259, c = idx - q * 259;
        int base = q * 24;
        float acc = 0.0f;
        if (c < 3) {
            for (int s = 0; s < 24; ++s) acc += wL[base + s] * rgbL[(base + s) * 3 + c];
            acc = -1.0f + 2.0f * acc;
        } else {
            int cc = c - 3;
            for (int s = 0; s < 24; ++s) acc += wL[base + s] * (float)buf1[(base + s) * SR + cc];
        }
        out[(size_t)(wg * 4 + q) * 259 + c] = acc;
    }
}

// ================= launcher =================
extern "C" void kernel_launch(void* const* d_in, const int* in_sizes, int n_in,
                              void* d_out, int out_size, void* d_ws, size_t ws_size,
                              hipStream_t stream)
{
    (void)in_sizes; (void)n_in; (void)out_size; (void)ws_size;
    const float* cam_poses   = (const float*)d_in[0];
    const float* focal       = (const float*)d_in[1];
    const float* nearv       = (const float*)d_in[2];
    const float* farv        = (const float*)d_in[3];
    const float* styles      = (const float*)d_in[4];
    const float* w0          = (const float*)d_in[5];
    const float* b0          = (const float*)d_in[6];
    const float* ws7         = (const float*)d_in[7];
    const float* bs          = (const float*)d_in[8];
    const float* gamma_w     = (const float*)d_in[9];
    const float* gamma_b     = (const float*)d_in[10];
    const float* beta_w      = (const float*)d_in[11];
    const float* beta_b      = (const float*)d_in[12];
    const float* view_w      = (const float*)d_in[13];
    const float* view_b      = (const float*)d_in[14];
    const float* view_gw     = (const float*)d_in[15];
    const float* view_gb     = (const float*)d_in[16];
    const float* view_bw     = (const float*)d_in[17];
    const float* view_bb     = (const float*)d_in[18];
    const float* rgb_w       = (const float*)d_in[19];
    const float* rgb_b       = (const float*)d_in[20];
    const float* sigma_w     = (const float*)d_in[21];
    const float* sigma_b     = (const float*)d_in[22];
    const float* sigmoid_beta= (const float*)d_in[23];

    _Float16* wh  = (_Float16*)d_ws;
    _Float16* w0h = wh;
    _Float16* wsh = wh + WSH_OFF;
    _Float16* vwh = wh + VWH_OFF;
    float* gammaArr = (float*)((char*)d_ws + GB_BYTE_OFF);
    float* betaArr  = gammaArr + 2 * 9 * 256;

    prep_weights<<<(F16_TOTAL + 255) / 256, 256, 0, stream>>>(w0, ws7, view_w, wh);
    prep_gb<<<(2 * 9 * 256 + 255) / 256, 256, 0, stream>>>(
        styles, gamma_w, gamma_b, beta_w, beta_b,
        view_gw, view_gb, view_bw, view_bb, gammaArr, betaArr);

    size_t smem = (size_t)WBUF_BYTES + 2 * (size_t)ACT_BYTES
                + (size_t)(NPTS + NPTS + NPTS * 3 + NPTS) * sizeof(float);
    (void)hipFuncSetAttribute((const void*)render_kernel,
                              hipFuncAttributeMaxDynamicSharedMemorySize, (int)smem);
    render_kernel<<<2048, NTHREADS, smem, stream>>>(
        cam_poses, focal, nearv, farv, b0, bs, view_b,
        rgb_w, rgb_b, sigma_w, sigma_b, sigmoid_beta,
        w0h, wsh, vwh, gammaArr, betaArr, (float*)d_out);
}
